// ManualDeepseekV2Attention_55619826483796
// MI455X (gfx1250) — matrix operand order repaired and checked by reading, not compiled, not run on hardware
//
#include <hip/hip_runtime.h>
#include <hip/hip_bf16.h>
#include <math.h>
#include <stdint.h>

// ---------------- problem constants ----------------
#define NHEAD   16
#define DNOPE   128
#define DROPE   64
#define DV      128
#define QLORA   1536
#define KVLORA  512
#define HID     2048
#define BB      2
#define SS      2048

typedef __bf16 bf16_t;
typedef __attribute__((ext_vector_type(16))) __bf16 v16bf;
typedef __attribute__((ext_vector_type(8)))  __bf16 v8bf;
typedef __attribute__((ext_vector_type(4)))  __bf16 v4bf;
typedef __attribute__((ext_vector_type(8)))  float  v8f;

__device__ __forceinline__ bf16_t f2bf(float f) {
  // round-to-nearest-even f32 -> bf16
  unsigned u = __builtin_bit_cast(unsigned, f);
  unsigned r = (u + 0x7FFFu + ((u >> 16) & 1u)) >> 16;
  unsigned short s = (unsigned short)r;
  return __builtin_bit_cast(bf16_t, s);
}

__device__ __forceinline__ v16bf cat8(v8bf lo, v8bf hi) {
  v16bf r;
#pragma unroll
  for (int i = 0; i < 8; ++i) { r[i] = lo[i]; r[i + 8] = hi[i]; }
  return r;
}

__device__ __forceinline__ v8f wmma_bf16(v16bf a, v16bf b, v8f c) {
  return __builtin_amdgcn_wmma_f32_16x16x32_bf16(
      false, a, false, b, (short)0, c, false, false);
}

// issue one 16-byte async global->LDS copy (ASYNCcnt-tracked, bypasses VGPRs)
__device__ __forceinline__ void async_b128(unsigned lds_addr, const void* gptr) {
  asm volatile("global_load_async_to_lds_b128 %0, %1, off"
               :: "v"(lds_addr), "v"(gptr) : "memory");
}
__device__ __forceinline__ unsigned lds_lo32(const void* p) {
  // generic LDS pointer low 32 bits == wave-relative LDS address
  return (unsigned)(uintptr_t)p;
}

// ============================================================
// elementwise f32 -> bf16 convert
// ============================================================
__global__ __launch_bounds__(256) void cvt_bf16_kernel(
    const float* __restrict__ in, bf16_t* __restrict__ out, int n) {
  int i = (blockIdx.x * 256 + threadIdx.x) * 4;
  if (i + 3 < n) {
    float4 v = *(const float4*)(in + i);
    v4bf o;
    o[0] = f2bf(v.x); o[1] = f2bf(v.y); o[2] = f2bf(v.z); o[3] = f2bf(v.w);
    *(v4bf*)(out + i) = o;
  } else {
    for (int j = 0; j < 4 && i + j < n; ++j) out[i + j] = f2bf(in[i + j]);
  }
}

// ============================================================
// transpose + convert: in[K][N] f32 -> out[N][K] bf16 (K,N % 32 == 0)
// ============================================================
__global__ __launch_bounds__(256) void transpose_cvt_kernel(
    const float* __restrict__ in, bf16_t* __restrict__ out, int K, int N) {
  __shared__ float tile[32][33];
  const int n0 = blockIdx.x * 32, k0 = blockIdx.y * 32;
  const int tx = threadIdx.x & 31, ty = threadIdx.x >> 5;  // 32 x 8
#pragma unroll
  for (int i = 0; i < 4; ++i) {
    int k = ty + i * 8;
    tile[k][tx] = in[(size_t)(k0 + k) * N + n0 + tx];
  }
  __syncthreads();
#pragma unroll
  for (int i = 0; i < 4; ++i) {
    int n = ty + i * 8;
    out[(size_t)(n0 + n) * K + k0 + tx] = f2bf(tile[tx][n]);
  }
}

// ============================================================
// GEMM: C[M,N](f32) = A[M,K](bf16) @ Bt[N,K](bf16, pre-transposed).
// 256 threads / 8 waves, 128x128 tile, K-step 32, double-buffered LDS,
// all staging via global_load_async_to_lds_b128 (software pipelined).
// ============================================================
__global__ __launch_bounds__(256) void gemm_bf16_wmma(
    const bf16_t* __restrict__ A, const bf16_t* __restrict__ Bt,
    float* __restrict__ C, int M, int N, int K) {
  __shared__ bf16_t As[2][128][32];   // [buf][m][k]
  __shared__ bf16_t Bs[2][128][32];   // [buf][n][k]

  const int tid = threadIdx.x;
  const int m0 = blockIdx.y * 128, n0 = blockIdx.x * 128;
  const int w = tid >> 5, l = tid & 31;
  const int wm = w >> 2, wn = w & 3;           // 2 x 4 wave grid
  const int hi = l >> 4, li = l & 15;
  const int kbA = hi * 8;                      // A operand K base
  const int kbB = hi * 8;                      // B operand K base

  v8f acc[4][2];
#pragma unroll
  for (int i = 0; i < 4; ++i)
#pragma unroll
    for (int j = 0; j < 2; ++j)
#pragma unroll
      for (int r = 0; r < 8; ++r) acc[i][j][r] = 0.f;

  // stage one 128x32 A tile + 128x32 B tile: 4 async b128 per thread
  auto stage = [&](int k0, int buf) {
    unsigned la = lds_lo32(&As[buf][0][0]);
    unsigned lb = lds_lo32(&Bs[buf][0][0]);
#pragma unroll
    for (int i = 0; i < 2; ++i) {
      int chunk = tid + i * 256;               // 512 chunks of 8 bf16
      int row = chunk >> 2;
      int kc = (chunk & 3) << 3;
      int ra = m0 + row; if (ra >= M) ra = M - 1;   // clamp: fixed issue count
      int rb = n0 + row; if (rb >= N) rb = N - 1;
      async_b128(la + (unsigned)((row * 32 + kc) * 2),
                 A + (size_t)ra * K + k0 + kc);
      async_b128(lb + (unsigned)((row * 32 + kc) * 2),
                 Bt + (size_t)rb * K + k0 + kc);
    }
  };

  const int nk = K >> 5;
  stage(0, 0);
  for (int t = 0; t < nk; ++t) {
    const int buf = t & 1;
    if (t + 1 < nk) {
      stage((t + 1) << 5, buf ^ 1);
      asm volatile("s_wait_asynccnt 0x4" ::: "memory");  // drain tile t (in-order)
    } else {
      asm volatile("s_wait_asynccnt 0x0" ::: "memory");
    }
    __syncthreads();

    v16bf av[4], bv[2];
#pragma unroll
    for (int i = 0; i < 4; ++i) {
      int row = wm * 64 + i * 16 + li;
      v8bf lo = *(const v8bf*)(&As[buf][row][kbA]);
      v8bf h8 = *(const v8bf*)(&As[buf][row][kbA + 16]);
      av[i] = cat8(lo, h8);
    }
#pragma unroll
    for (int j = 0; j < 2; ++j) {
      int col = wn * 32 + j * 16 + li;
      v8bf lo = *(const v8bf*)(&Bs[buf][col][kbB]);
      v8bf h8 = *(const v8bf*)(&Bs[buf][col][kbB + 16]);
      bv[j] = cat8(lo, h8);
    }
#pragma unroll
    for (int i = 0; i < 4; ++i)
#pragma unroll
      for (int j = 0; j < 2; ++j)
        acc[i][j] = wmma_bf16(av[i], bv[j], acc[i][j]);
    __syncthreads();
  }

#pragma unroll
  for (int i = 0; i < 4; ++i)
#pragma unroll
    for (int j = 0; j < 2; ++j)
#pragma unroll
      for (int r = 0; r < 8; ++r) {
        int gm = m0 + wm * 64 + i * 16 + r + hi * 8;
        int gn = n0 + wn * 32 + j * 16 + li;
        if (gm < M && gn < N) C[(size_t)gm * N + gn] = acc[i][j][r];
      }
}

// ============================================================
// RMSNorm: f32 in -> bf16 out, one block per row.
// ============================================================
__global__ __launch_bounds__(256) void rmsnorm_kernel(
    const float* __restrict__ in, int istride, int n,
    const float* __restrict__ w, bf16_t* __restrict__ out, int ostride) {
  __shared__ float red[256];
  const int row = blockIdx.x, tid = threadIdx.x;
  const float* x = in + (size_t)row * istride;
  float ss = 0.f;
  for (int i = tid; i < n; i += 256) { float v = x[i]; ss += v * v; }
  red[tid] = ss;
  __syncthreads();
  for (int s = 128; s > 0; s >>= 1) {
    if (tid < s) red[tid] += red[tid + s];
    __syncthreads();
  }
  float sc = rsqrtf(red[0] / (float)n + 1e-6f);
  bf16_t* o = out + (size_t)row * ostride;
  for (int i = tid; i < n; i += 256) o[i] = f2bf(w[i] * x[i] * sc);
}

// ============================================================
// RoPE (interleaved) + pack: Q/K -> bf16 [b,h,s,d]; V -> bf16 [b,h,dv,s]
// (V pre-transposed so attention stages it as a linear async copy)
// ============================================================
__global__ __launch_bounds__(192) void rope_pack_kernel(
    const float* __restrict__ q,    // [B*S, 16*192]
    const float* __restrict__ kv,   // [B*S, 16*256]
    const float* __restrict__ ckv,  // [B*S, 576]
    const float* __restrict__ cosp, const float* __restrict__ sinp,
    bf16_t* __restrict__ Qb, bf16_t* __restrict__ Kb, bf16_t* __restrict__ Vbt) {
  int idx = blockIdx.x;
  int h = idx & 15;
  int s = (idx >> 4) & (SS - 1);
  int b = idx >> 15;
  int d = threadIdx.x;
  size_t bs = (size_t)b * SS + s;
  size_t bh = (size_t)(b * NHEAD + h);
  size_t bhs = bh * SS + s;

  float qv;
  if (d < DNOPE) {
    qv = q[bs * (NHEAD * 192) + h * 192 + d];
  } else {
    int j = d - DNOPE, p = j >> 1;
    float cv = cosp[(size_t)s * DROPE + 2 * p];
    float sv = sinp[(size_t)s * DROPE + 2 * p];
    float xr = q[bs * (NHEAD * 192) + h * 192 + DNOPE + 2 * p];
    float xi = q[bs * (NHEAD * 192) + h * 192 + DNOPE + 2 * p + 1];
    qv = (j & 1) ? (xr * sv + xi * cv) : (xr * cv - xi * sv);
  }
  Qb[bhs * 192 + d] = f2bf(qv);

  float kx;
  if (d < DNOPE) {
    kx = kv[bs * (NHEAD * 256) + h * 256 + d];
  } else {
    int j = d - DNOPE, p = j >> 1;
    float cv = cosp[(size_t)s * DROPE + 2 * p];
    float sv = sinp[(size_t)s * DROPE + 2 * p];
    float xr = ckv[bs * 576 + KVLORA + 2 * p];
    float xi = ckv[bs * 576 + KVLORA + 2 * p + 1];
    kx = (j & 1) ? (xr * sv + xi * cv) : (xr * cv - xi * sv);
  }
  Kb[bhs * 192 + d] = f2bf(kx);

  if (d < DV)
    Vbt[(bh * DV + d) * SS + s] = f2bf(kv[bs * (NHEAD * 256) + h * 256 + DNOPE + d]);
}

// ============================================================
// Flash attention: grid (16 q-tiles, 32 b*h), 256 threads, key tiles of 64.
// K / Vt tiles staged via async global->LDS; bf16 output for o-proj.
// ============================================================
__global__ __launch_bounds__(256) void attn_kernel(
    const bf16_t* __restrict__ Qb, const bf16_t* __restrict__ Kb,
    const bf16_t* __restrict__ Vbt, bf16_t* __restrict__ out, float scale) {
  __shared__ bf16_t Ks[64][192];       // [key][d]
  __shared__ bf16_t Vt[128][64];       // [dv][key]
  __shared__ bf16_t Pb[8][16][64];     // per-wave P scratch [m][key]

  const int tid = threadIdx.x;
  const int qt = blockIdx.x;           // 0..15
  const int bh = blockIdx.y;           // 0..31
  const int w = tid >> 5, l = tid & 31;
  const int hi = l >> 4, li = l & 15;
  const int kbA = hi * 8, kbB = hi * 8;
  const int q0 = qt * 128 + w * 16;

  const unsigned ldsK = lds_lo32(&Ks[0][0]);
  const unsigned ldsV = lds_lo32(&Vt[0][0]);

  // preload this wave's Q operands (16 rows x 192) into registers
  const bf16_t* Qrow = Qb + ((size_t)bh * SS + q0 + li) * 192;
  v16bf qa[6];
#pragma unroll
  for (int c = 0; c < 6; ++c) {
    v8bf lo = *(const v8bf*)(Qrow + c * 32 + kbA);
    v8bf h8 = *(const v8bf*)(Qrow + c * 32 + kbA + 16);
    qa[c] = cat8(lo, h8);
  }

  v8f O[8];
  float mrow[8], lrow[8];
#pragma unroll
  for (int r = 0; r < 8; ++r) { mrow[r] = -3.0e38f; lrow[r] = 0.f; }
#pragma unroll
  for (int nb = 0; nb < 8; ++nb)
#pragma unroll
    for (int r = 0; r < 8; ++r) O[nb][r] = 0.f;

  for (int kt = 0; kt < SS / 64; ++kt) {
    // stage K tile (64x192 bf16, linear): 6 async b128 per thread
    {
      const bf16_t* src = Kb + ((size_t)bh * SS + kt * 64) * 192;
#pragma unroll
      for (int i = 0; i < 6; ++i) {
        int e = (tid + i * 256) * 8;
        async_b128(ldsK + (unsigned)(e * 2), src + e);
      }
    }
    // stage Vt tile (128 dv-rows x 64 keys): 4 async b128 per thread
    {
      const bf16_t* src = Vbt + (size_t)bh * DV * SS + kt * 64;
#pragma unroll
      for (int i = 0; i < 4; ++i) {
        int chunk = tid + i * 256;       // 1024 chunks of 8 bf16
        int dv = chunk >> 3;
        int kc = (chunk & 7) << 3;
        async_b128(ldsV + (unsigned)((dv * 64 + kc) * 2),
                   src + (size_t)dv * SS + kc);
      }
    }
    asm volatile("s_wait_asynccnt 0x0" ::: "memory");
    __syncthreads();

    // S = Q K^T  (4 blocks of 16 keys, contraction over d=192)
    v8f sacc[4];
#pragma unroll
    for (int nb = 0; nb < 4; ++nb) {
#pragma unroll
      for (int r = 0; r < 8; ++r) sacc[nb][r] = 0.f;
#pragma unroll
      for (int c = 0; c < 6; ++c) {
        const bf16_t* kr = &Ks[nb * 16 + li][c * 32 + kbB];
        v8bf blo = *(const v8bf*)(kr);
        v8bf bhi = *(const v8bf*)(kr + 16);
        sacc[nb] = wmma_bf16(qa[c], cat8(blo, bhi), sacc[nb]);
      }
    }

    // online softmax (rows live in 16-lane halves; xor-shuffle reduce)
    float tmax[8];
#pragma unroll
    for (int r = 0; r < 8; ++r) tmax[r] = -3.0e38f;
#pragma unroll
    for (int nb = 0; nb < 4; ++nb)
#pragma unroll
      for (int r = 0; r < 8; ++r) tmax[r] = fmaxf(tmax[r], sacc[nb][r] * scale);
#pragma unroll
    for (int off = 1; off < 16; off <<= 1)
#pragma unroll
      for (int r = 0; r < 8; ++r)
        tmax[r] = fmaxf(tmax[r], __shfl_xor(tmax[r], off, 32));

    float fac[8];
#pragma unroll
    for (int r = 0; r < 8; ++r) {
      float mn = fmaxf(mrow[r], tmax[r]);
      fac[r] = __expf(mrow[r] - mn);
      mrow[r] = mn;
    }
    float rsum[8];
#pragma unroll
    for (int r = 0; r < 8; ++r) rsum[r] = 0.f;
#pragma unroll
    for (int nb = 0; nb < 4; ++nb)
#pragma unroll
      for (int r = 0; r < 8; ++r) {
        float p = __expf(sacc[nb][r] * scale - mrow[r]);
        rsum[r] += p;
        Pb[w][r + hi * 8][nb * 16 + li] = f2bf(p);
      }
#pragma unroll
    for (int off = 1; off < 16; off <<= 1)
#pragma unroll
      for (int r = 0; r < 8; ++r) rsum[r] += __shfl_xor(rsum[r], off, 32);
#pragma unroll
    for (int r = 0; r < 8; ++r) lrow[r] = lrow[r] * fac[r] + rsum[r];
#pragma unroll
    for (int nb = 0; nb < 8; ++nb)
#pragma unroll
      for (int r = 0; r < 8; ++r) O[nb][r] *= fac[r];

    // ensure P stores complete before operand re-load (per-wave LDS order)
    asm volatile("s_wait_dscnt 0" ::: "memory");

    // O += P V  (contraction over 64 keys, 2 chunks of 32)
#pragma unroll
    for (int c = 0; c < 2; ++c) {
      v8bf plo = *(const v8bf*)(&Pb[w][li][c * 32 + kbA]);
      v8bf phi = *(const v8bf*)(&Pb[w][li][c * 32 + kbA + 16]);
      v16bf pa = cat8(plo, phi);
#pragma unroll
      for (int nb = 0; nb < 8; ++nb) {
        const bf16_t* vr = &Vt[nb * 16 + li][c * 32 + kbB];
        v8bf vlo = *(const v8bf*)(vr);
        v8bf vhi = *(const v8bf*)(vr + 16);
        O[nb] = wmma_bf16(pa, cat8(vlo, vhi), O[nb]);
      }
    }
    __syncthreads();
  }

  // epilogue: bf16 out[b, q, h*128 + dv] feeds the o-proj GEMM directly
  const int b = bh >> 4, h = bh & 15;
#pragma unroll
  for (int nb = 0; nb < 8; ++nb)
#pragma unroll
    for (int r = 0; r < 8; ++r) {
      int m = r + hi * 8;
      int dv = nb * 16 + li;
      out[((size_t)b * SS + q0 + m) * (NHEAD * DV) + h * DV + dv] =
          f2bf(O[nb][r] / lrow[r]);
    }
}

// ============================================================
// host launcher
// ============================================================
extern "C" void kernel_launch(void* const* d_in, const int* in_sizes, int n_in,
                              void* d_out, int out_size, void* d_ws, size_t ws_size,
                              hipStream_t stream) {
  (void)in_sizes; (void)n_in; (void)out_size; (void)ws_size;
  const float* hidden  = (const float*)d_in[0];
  const float* cosp    = (const float*)d_in[1];
  const float* sinp    = (const float*)d_in[2];
  const float* q_a_w   = (const float*)d_in[3];
  const float* q_a_ln  = (const float*)d_in[4];
  const float* q_b_w   = (const float*)d_in[5];
  const float* kv_a_w  = (const float*)d_in[6];
  const float* kv_a_ln = (const float*)d_in[7];
  const float* kv_b_w  = (const float*)d_in[8];
  const float* o_w     = (const float*)d_in[9];
  float* outp = (float*)d_out;

  char* ws = (char*)d_ws;
  size_t off = 0;
  auto alloc = [&](size_t bytes) -> void* {
    void* p = ws + off;
    off += (bytes + 255) & ~(size_t)255;
    return p;
  };
  const int MS = BB * SS;  // 4096 rows
  // bf16 operand buffers
  bf16_t* hid_bf = (bf16_t*)alloc((size_t)MS * HID * 2);
  bf16_t* qaw_t  = (bf16_t*)alloc((size_t)QLORA * HID * 2);
  bf16_t* qbw_t  = (bf16_t*)alloc((size_t)(NHEAD * 192) * QLORA * 2);
  bf16_t* kvaw_t = (bf16_t*)alloc((size_t)(KVLORA + DROPE) * HID * 2);
  bf16_t* kvbw_t = (bf16_t*)alloc((size_t)(NHEAD * 256) * KVLORA * 2);
  bf16_t* ow_t   = (bf16_t*)alloc((size_t)HID * (NHEAD * DV) * 2);
  // f32 GEMM outputs
  float*  q_a    = (float*)alloc((size_t)MS * QLORA * 4);
  float*  qmat   = (float*)alloc((size_t)MS * (NHEAD * 192) * 4);
  float*  ckv    = (float*)alloc((size_t)MS * 576 * 4);
  float*  kvmat  = (float*)alloc((size_t)MS * (NHEAD * 256) * 4);
  // bf16 normalized activations
  bf16_t* qan_bf = (bf16_t*)alloc((size_t)MS * QLORA * 2);
  bf16_t* ckvn_bf= (bf16_t*)alloc((size_t)MS * KVLORA * 2);
  // attention operands / output
  bf16_t* Qb     = (bf16_t*)alloc((size_t)BB * NHEAD * SS * 192 * 2);
  bf16_t* Kb     = (bf16_t*)alloc((size_t)BB * NHEAD * SS * 192 * 2);
  bf16_t* Vbt    = (bf16_t*)alloc((size_t)BB * NHEAD * DV * SS * 2);
  bf16_t* attnbf = (bf16_t*)alloc((size_t)MS * (NHEAD * DV) * 2);

  auto cvt = [&](const float* in, bf16_t* out, size_t n) {
    cvt_bf16_kernel<<<(unsigned)((n / 4 + 255) / 256), 256, 0, stream>>>(in, out, (int)n);
  };
  auto tcvt = [&](const float* in, bf16_t* out, int K, int N) {
    transpose_cvt_kernel<<<dim3(N / 32, K / 32), 256, 0, stream>>>(in, out, K, N);
  };
  auto gemm = [&](const bf16_t* A, const bf16_t* Bt, float* C, int M, int N, int K) {
    dim3 g((N + 127) / 128, (M + 127) / 128);
    gemm_bf16_wmma<<<g, 256, 0, stream>>>(A, Bt, C, M, N, K);
  };

  // one-time conversions (O(bytes), off the critical FLOP path)
  cvt(hidden, hid_bf, (size_t)MS * HID);
  tcvt(q_a_w, qaw_t, HID, QLORA);
  tcvt(q_b_w, qbw_t, QLORA, NHEAD * 192);
  tcvt(kv_a_w, kvaw_t, HID, KVLORA + DROPE);
  tcvt(kv_b_w, kvbw_t, KVLORA, NHEAD * 256);
  tcvt(o_w, ow_t, NHEAD * DV, HID);

  gemm(hid_bf, qaw_t, q_a, MS, QLORA, HID);
  rmsnorm_kernel<<<MS, 256, 0, stream>>>(q_a, QLORA, QLORA, q_a_ln, qan_bf, QLORA);
  gemm(qan_bf, qbw_t, qmat, MS, NHEAD * 192, QLORA);
  gemm(hid_bf, kvaw_t, ckv, MS, KVLORA + DROPE, HID);
  rmsnorm_kernel<<<MS, 256, 0, stream>>>(ckv, KVLORA + DROPE, KVLORA, kv_a_ln, ckvn_bf, KVLORA);
  gemm(ckvn_bf, kvbw_t, kvmat, MS, NHEAD * 256, KVLORA);
  rope_pack_kernel<<<BB * SS * NHEAD, 192, 0, stream>>>(qmat, kvmat, ckv, cosp, sinp, Qb, Kb, Vbt);
  const float scale = 0.07216878364870323f;  // 1/sqrt(192)
  attn_kernel<<<dim3(SS / 128, BB * NHEAD), 256, 0, stream>>>(Qb, Kb, Vbt, attnbf, scale);
  gemm(attnbf, ow_t, outp, MS, HID, NHEAD * DV);
}
